// SmartPool2d_4234837754578
// MI455X (gfx1250) — matrix operand order repaired and checked by reading, not compiled, not run on hardware
//
#include <hip/hip_runtime.h>

#define N_ 32
#define C_ 128
#define H_ 128
#define W_ 128
#define OH_ 64
#define OW_ 64
#define RATIO_ 0.1f

typedef __attribute__((ext_vector_type(16))) __bf16 v16bf;
typedef __attribute__((ext_vector_type(8)))  float  v8f;
typedef __attribute__((ext_vector_type(4)))  unsigned int v4u;
typedef __attribute__((ext_vector_type(8)))  int    v8i;
typedef __attribute__((ext_vector_type(4)))  int    v4i;

union Frag { uint4 u[2]; v16bf v; };
union Acc  { v8f v; float f[8]; };

// ---- dynamic LDS layout (bytes) ----
#define OFF_X    0        // float[128*128]      65536
#define OFF_XT   65536    // ushort[128*136]     34816  (x transposed, bf16, padded pitch)
#define OFF_WH   100352   // ushort[64*128]      16384
#define OFF_WW   116736   // ushort[64*128]      16384
#define OFF_TMP  133120   // ushort[64*128]      16384
#define OFF_RED  149504   // float[256]           1024
#define OFF_RED4 150528   // float4[256]          4096
#define SMEM_BYTES 154624
#define XT_PITCH 136

__device__ __forceinline__ unsigned short f2bf(float f) {
  unsigned int u = __float_as_uint(f);
  u += 0x7FFFu + ((u >> 16) & 1u);     // round-to-nearest-even
  return (unsigned short)(u >> 16);
}

// two f32 -> packed bf16 dword (hardware cvt when available)
__device__ __forceinline__ unsigned int pk_bf16(float a, float b) {
#if __has_builtin(__builtin_amdgcn_cvt_pk_bf16_f32)
  auto r = __builtin_amdgcn_cvt_pk_bf16_f32(a, b);
  unsigned int u;
  __builtin_memcpy(&u, &r, 4);
  return u;
#else
  return (unsigned int)f2bf(a) | ((unsigned int)f2bf(b) << 16);
#endif
}

// Antialiased triangle resize weights, one 64x128 matrix, row-normalized.
// 4 lanes cooperate on one output row (row sum via wave32 shuffles).
__device__ __forceinline__ void gen_weights(float lo, float hi,
                                            unsigned short* dst, int tid) {
  const float L = hi - lo + 1.0f;
  const float scale = L * (1.0f / 64.0f);
  const float support = fmaxf(scale, 1.0f);
  const float inv_support = 1.0f / support;
  const int o = tid >> 2;
  const int q = tid & 3;
  const float center = scale * ((float)o + 0.5f);
  float wv[32];
  float s = 0.0f;
#pragma unroll
  for (int jj = 0; jj < 32; ++jj) {
    const float j = (float)(q * 32 + jj);
    const float pos = j - lo + 0.5f;
    const float t = (pos - center) * inv_support;
    float w = fmaxf(0.0f, 1.0f - fabsf(t));
    w = ((j >= lo) && (j <= hi)) ? w : 0.0f;
    wv[jj] = w;
    s += w;
  }
  s += __shfl_xor(s, 1, 32);
  s += __shfl_xor(s, 2, 32);
  const float inv = 1.0f / fmaxf(s, 1e-12f);
#pragma unroll
  for (int jj = 0; jj < 32; jj += 2)
    *(unsigned int*)(dst + o * 128 + q * 32 + jj) =
        pk_bf16(wv[jj] * inv, wv[jj + 1] * inv);
}

__global__ __launch_bounds__(256)
void smartpool2d_kernel(const float* __restrict__ x, float* __restrict__ out) {
  extern __shared__ char smem[];
  float*          lds_x  = (float*)(smem + OFF_X);
  unsigned short* lds_xt = (unsigned short*)(smem + OFF_XT);
  unsigned short* lds_wh = (unsigned short*)(smem + OFF_WH);
  unsigned short* lds_ww = (unsigned short*)(smem + OFF_WW);
  unsigned short* lds_tp = (unsigned short*)(smem + OFF_TMP);
  float*          red    = (float*)(smem + OFF_RED);
  float4*         red4   = (float4*)(smem + OFF_RED4);

  const int tid = threadIdx.x;
  const int nc  = blockIdx.x;
  const float* src = x + (size_t)nc * (H_ * W_);

  // ---- Stage 0: TDM: DMA the 128x128 f32 plane into LDS (wave 0 issues) ----
  if (tid < 32) {
    const unsigned long long ga = (unsigned long long)(const void*)src;
    const unsigned int ldsa = (unsigned int)(size_t)(void*)lds_x;
    // D# group0: count=1 | lds_addr | global_addr[56:0] | type=2
    v4u g0 = { 1u, ldsa,
               (unsigned int)(ga & 0xFFFFFFFFull),
               (unsigned int)(((ga >> 32) & 0x1FFFFFFull) | (2u << 30)) };
    // D# group1: data_size=4B, tensor 128x128, tile 128x128, strides 128 / 16384
    v8i g1 = { (int)(2u << 16),
               (int)(((unsigned)W_) << 16),                 // tensor_dim0 lo16
               (int)(((unsigned)H_) << 16),                 // tensor_dim1 lo16
               (int)(((unsigned)W_) << 16),                 // tile_dim0
               (int)H_,                                     // tile_dim1
               (int)W_,                                     // tensor_dim0_stride lo32
               (int)((((unsigned)(H_ * W_)) & 0xFFFFu) << 16), // dim1_stride lo16
               0 };
    v4i gz4 = { 0, 0, 0, 0 };
    v8i gz8 = { 0, 0, 0, 0, 0, 0, 0, 0 };
    __builtin_amdgcn_tensor_load_to_lds(g0, g1, gz4, gz4, gz8, 0);
    __builtin_amdgcn_s_wait_tensorcnt(0);
  }
  __syncthreads();

  const float4* xv = (const float4*)lds_x;   // 4096 x float4

  // ---- Stage 1: plane max (vectorized) ----
  float m = -3.4e38f;
  for (int e = tid; e < (H_ * W_) / 4; e += 256) {
    const float4 v = xv[e];
    m = fmaxf(m, fmaxf(fmaxf(v.x, v.y), fmaxf(v.z, v.w)));
  }
  red[tid] = m;
  __syncthreads();
  for (int s = 128; s > 0; s >>= 1) {
    if (tid < s) red[tid] = fmaxf(red[tid], red[tid + s]);
    __syncthreads();
  }
  const float thr = RATIO_ * red[0];

  // ---- Stage 2: bbox reductions (group-folded, identical to reference) ----
  // The +/-1e5 sentinels at h==0 / w==0 can never win the max reductions
  // (non-negative candidates always exist for h,w>0), so those selects fold.
  float xmn = 3.4e38f, xmx = -3.4e38f, ymn = 3.4e38f, ymx = -3.4e38f;
  for (int e = tid; e < (H_ * W_) / 4; e += 256) {
    const int h  = e >> 5;            // 32 float4 per row
    const int w0 = (e & 31) * 4;
    const float4 v = xv[e];
    const bool t0 = v.x >= thr, t1 = v.y >= thr, t2 = v.z >= thr, t3 = v.w >= thr;
    const bool anyT = t0 || t1 || t2 || t3;
    const bool allT = t0 && t1 && t2 && t3;
    const float fh = (float)h;
    xmx = fmaxf(xmx, anyT ? fh : 0.0f);
    xmn = fminf(xmn, (h == 0) ? 1e5f : (allT ? fh : 0.0f));
    const float fw1 = t1 ? (float)(w0 + 1) : 0.0f;
    const float fw2 = t2 ? (float)(w0 + 2) : 0.0f;
    const float fw3 = t3 ? (float)(w0 + 3) : 0.0f;
    const float fw0 = t0 ? (float)w0 : 0.0f;
    ymx = fmaxf(ymx, fmaxf(fmaxf(fw0, fw1), fmaxf(fw2, fw3)));
    const float fw0m = (w0 == 0) ? 1e5f : fw0;
    ymn = fminf(ymn, fminf(fminf(fw0m, fw1), fminf(fw2, fw3)));
  }
  red4[tid] = make_float4(xmn, xmx, ymn, ymx);
  __syncthreads();
  for (int s = 128; s > 0; s >>= 1) {
    if (tid < s) {
      float4 a = red4[tid], b = red4[tid + s];
      a.x = fminf(a.x, b.x); a.y = fmaxf(a.y, b.y);
      a.z = fminf(a.z, b.z); a.w = fmaxf(a.w, b.w);
      red4[tid] = a;
    }
    __syncthreads();
  }
  const float4 bb = red4[0];

  // ---- Stage 3: per-channel resize weights (bf16 in LDS) ----
  gen_weights(bb.x, bb.y, lds_wh, tid);   // Wh[o][h]  (A, row-major)
  gen_weights(bb.z, bb.w, lds_ww, tid);   // Ww[p][w]  (== n-major B for GEMM2)

  // ---- Stage 4: x -> bf16 transposed [w][h], packed dword stores ----
  for (int e = tid; e < (H_ * W_) / 2; e += 256) {
    const int w  = e & 127;
    const int h2 = (e >> 7) << 1;        // even h
    const float a = lds_x[h2 * W_ + w];
    const float b = lds_x[(h2 + 1) * W_ + w];
    *(unsigned int*)(lds_xt + w * XT_PITCH + h2) = pk_bf16(a, b);
  }
  __syncthreads();

  const int wv = tid >> 5, lane = tid & 31;
  const int nl = lane & 15, kh = lane >> 4;

  // ---- Stage 5: GEMM1  tmp[64x128] = Wh @ x  (K = h = 128) ----
  {
    Acc acc[4];
#pragma unroll
    for (int mt = 0; mt < 4; ++mt)
#pragma unroll
      for (int i = 0; i < 8; ++i) acc[mt].f[i] = 0.0f;
    const int n0 = wv * 16;               // each wave owns one 16-wide n column
#pragma unroll
    for (int kb = 0; kb < 4; ++kb) {
      Frag b;                              // B frag: 32 contiguous bytes in xT
      const uint4* bp =
          (const uint4*)(lds_xt + (n0 + nl) * XT_PITCH + kb * 32 + kh * 16);
      b.u[0] = bp[0];
      b.u[1] = bp[1];
#pragma unroll
      for (int mt = 0; mt < 4; ++mt) {
        Frag a;                            // A frag: 32 contiguous bytes in Wh
        const uint4* ap =
            (const uint4*)(lds_wh + (mt * 16 + nl) * 128 + kb * 32 + kh * 16);
        a.u[0] = ap[0];
        a.u[1] = ap[1];
        acc[mt].v = __builtin_amdgcn_wmma_f32_16x16x32_bf16(
            false, a.v, false, b.v, (short)0, acc[mt].v, false, false);
      }
    }
#pragma unroll
    for (int mt = 0; mt < 4; ++mt)
#pragma unroll
      for (int r = 0; r < 8; ++r) {
        const int row = mt * 16 + (kh << 3) + r;   // D layout: VGPR r -> M=r / 8+r
        lds_tp[row * 128 + n0 + nl] = f2bf(acc[mt].f[r]);
      }
  }
  __syncthreads();

  // ---- Stage 6: GEMM2  out[64x64] = tmp @ Ww^T  (K = w = 128) ----
  {
    const int m0 = (wv & 3) * 16;
    const int n0 = (wv >> 2) * 32;        // each wave owns two adjacent n tiles
    Acc c0, c1;
#pragma unroll
    for (int i = 0; i < 8; ++i) { c0.f[i] = 0.0f; c1.f[i] = 0.0f; }
#pragma unroll
    for (int kb = 0; kb < 4; ++kb) {
      Frag a;
      const uint4* ap =
          (const uint4*)(lds_tp + (m0 + nl) * 128 + kb * 32 + kh * 16);
      a.u[0] = ap[0];
      a.u[1] = ap[1];
      Frag b0, b1;
      const uint4* b0p =
          (const uint4*)(lds_ww + (n0 + nl) * 128 + kb * 32 + kh * 16);
      const uint4* b1p =
          (const uint4*)(lds_ww + (n0 + 16 + nl) * 128 + kb * 32 + kh * 16);
      b0.u[0] = b0p[0]; b0.u[1] = b0p[1];
      b1.u[0] = b1p[0]; b1.u[1] = b1p[1];
      c0.v = __builtin_amdgcn_wmma_f32_16x16x32_bf16(
          false, a.v, false, b0.v, (short)0, c0.v, false, false);
      c1.v = __builtin_amdgcn_wmma_f32_16x16x32_bf16(
          false, a.v, false, b1.v, (short)0, c1.v, false, false);
    }
    float* dst = out + (size_t)nc * (OH_ * OW_);
#pragma unroll
    for (int r = 0; r < 8; ++r) {
      const int row = m0 + (kh << 3) + r;
      dst[row * OW_ + n0 + nl]      = c0.f[r];
      dst[row * OW_ + n0 + 16 + nl] = c1.f[r];
    }
  }
}

extern "C" void kernel_launch(void* const* d_in, const int* in_sizes, int n_in,
                              void* d_out, int out_size, void* d_ws, size_t ws_size,
                              hipStream_t stream) {
  (void)in_sizes; (void)n_in; (void)d_ws; (void)ws_size; (void)out_size;
  const float* x = (const float*)d_in[0];
  float* out = (float*)d_out;
  (void)hipFuncSetAttribute((const void*)smartpool2d_kernel,
                            hipFuncAttributeMaxDynamicSharedMemorySize,
                            SMEM_BYTES);
  smartpool2d_kernel<<<N_ * C_, 256, SMEM_BYTES, stream>>>(x, out);
}
